// Gemma4AudioAttention_5617817223257
// MI455X (gfx1250) — matrix operand order repaired and checked by reading, not compiled, not run on hardware
//
#include <hip/hip_runtime.h>
#include <hip/hip_bf16.h>
#include <stdint.h>

// ---------------------------------------------------------------------------
// Gemma4 audio attention on gfx1250 (MI455X), bf16 WMMA pipeline:
//   1) cvt_hidden      : f32 hidden -> bf16
//   2) cvt_weights     : f32 {w_q*qscale, w_k*K_SCALE, w_v, w_post} -> bf16
//   3) relk_kernel     : rel_k = pos_emb @ w_rel^T  (tiny)
//   4) gemm_bf16<true> : QKV = hidden @ [wq|wk|wv]^T (32768x3072x1024) -> bf16
//   5) attn_kernel     : per (b, chunk, head): WMMA scores, rel-shift,
//                        tanh softcap, softmax, attn@v via WMMA -> bf16
//   6) gemm_bf16<false>: out = attn_out @ w_post^T -> f32
// Guards are replaced by clamped addresses; zero-padding semantics are
// re-applied in the softmax (scores) and attention-weight (V) stages.
// ---------------------------------------------------------------------------

typedef __bf16 bf16_t;
typedef __attribute__((ext_vector_type(16))) __bf16 v16bf;
typedef __attribute__((ext_vector_type(8)))  float  v8f;

union Frag {
    v16bf    v;
    uint32_t u[8];
};
union BfPair {
    uint32_t u;
    bf16_t   h[2];
};

__device__ __forceinline__ bf16_t f2bf(float f) { return (bf16_t)f; }

__device__ __forceinline__ v8f zero8() {
    v8f z = {0.f, 0.f, 0.f, 0.f, 0.f, 0.f, 0.f, 0.f};
    return z;
}

// Fragment = two 16-byte contiguous halves (per-lane layout of CDNA5 WMMA bf16).
__device__ __forceinline__ Frag load_frag2(const bf16_t* p0, const bf16_t* p1) {
    Frag f;
    uint4 lo = *(const uint4*)p0;
    uint4 hi = *(const uint4*)p1;
    f.u[0] = lo.x; f.u[1] = lo.y; f.u[2] = lo.z; f.u[3] = lo.w;
    f.u[4] = hi.x; f.u[5] = hi.y; f.u[6] = hi.z; f.u[7] = hi.w;
    return f;
}

__device__ __forceinline__ int clampi(int v, int lo, int hi) {
    return v < lo ? lo : (v > hi ? hi : v);
}

#define WMMA_BF16(a, b, c) \
    __builtin_amdgcn_wmma_f32_16x16x32_bf16(false, (a), false, (b), (short)0, (c), false, false)

// ---------------------------------------------------------------------------
// 1) hidden f32 -> bf16
// ---------------------------------------------------------------------------
__global__ __launch_bounds__(256)
void cvt_hidden(const float* __restrict__ src, bf16_t* __restrict__ dst) {
    size_t i = ((size_t)blockIdx.x * 256 + threadIdx.x) * 4;
    float4 f = *(const float4*)(src + i);
    dst[i + 0] = f2bf(f.x);
    dst[i + 1] = f2bf(f.y);
    dst[i + 2] = f2bf(f.z);
    dst[i + 3] = f2bf(f.w);
}

// ---------------------------------------------------------------------------
// 2) weights f32 -> bf16 with folded scales (rows: wq | wk | wv | wpost)
// ---------------------------------------------------------------------------
__global__ __launch_bounds__(256)
void cvt_weights(const float* __restrict__ wq, const float* __restrict__ wk,
                 const float* __restrict__ wv, const float* __restrict__ wp,
                 const float* __restrict__ pds, bf16_t* __restrict__ dst) {
    size_t i = ((size_t)blockIdx.x * 256 + threadIdx.x) * 4;
    int row = (int)(i >> 10);
    int col = (int)(i & 1023);
    const float* src;
    float s;
    if (row < 1024) {
        src = wq + (size_t)row * 1024;
        s = 0.18033688f * log1pf(__expf(pds[row & 63]));   // Q_SCALE*softplus
    } else if (row < 2048) {
        src = wk + (size_t)(row - 1024) * 1024;
        s = 1.89468246f;                                    // K_SCALE
    } else if (row < 3072) {
        src = wv + (size_t)(row - 2048) * 1024;
        s = 1.0f;
    } else {
        src = wp + (size_t)(row - 3072) * 1024;
        s = 1.0f;
    }
    float4 f = *(const float4*)(src + col);
    dst[i + 0] = f2bf(f.x * s);
    dst[i + 1] = f2bf(f.y * s);
    dst[i + 2] = f2bf(f.z * s);
    dst[i + 3] = f2bf(f.w * s);
}

// ---------------------------------------------------------------------------
// 3) rel_k[l][n] = sum_k pos_emb[l][k] * w_rel[n][k]
// ---------------------------------------------------------------------------
__global__ __launch_bounds__(256)
void relk_kernel(const float* __restrict__ pe, const float* __restrict__ wr,
                 bf16_t* __restrict__ dst) {
    int idx = blockIdx.x * 256 + threadIdx.x;
    if (idx >= 71 * 1024) return;
    int l = idx >> 10, n = idx & 1023;
    const float* a = pe + (size_t)l * 1024;
    const float* b = wr + (size_t)n * 1024;
    float acc = 0.f;
    for (int k = 0; k < 1024; ++k) acc += a[k] * b[k];
    dst[idx] = f2bf(acc);
}

// ---------------------------------------------------------------------------
// 4/6) C[M][N] = A[M][K] * B[N][K]^T.  256 thr = 8 waves, tile 128x64.
// ---------------------------------------------------------------------------
template <bool OUT_BF16>
__global__ __launch_bounds__(256)
void gemm_bf16_kernel(const bf16_t* __restrict__ A, const bf16_t* __restrict__ B,
                      void* __restrict__ Cout, int M, int N, int K) {
    __shared__ bf16_t As[128][32];
    __shared__ bf16_t Bs[64][32];
    const int tid  = threadIdx.x;
    const int lane = tid & 31;
    const int wid  = tid >> 5;
    const int m0 = blockIdx.x * 128;
    const int n0 = blockIdx.y * 64;
    const int wm = (wid & 3) * 32;
    const int wn = (wid >> 2) * 32;
    const int g = lane >> 4;
    const int x = lane & 15;

    const int rA0 = tid >> 2,         cA0 = (tid & 3) * 8;
    const int rA1 = (tid + 256) >> 2, cA1 = cA0;
    const int rB  = tid >> 2,         cB  = (tid & 3) * 8;

    v8f c[2][2];
    c[0][0] = zero8(); c[0][1] = zero8();
    c[1][0] = zero8(); c[1][1] = zero8();

    for (int kk = 0; kk < K; kk += 32) {
        *(uint4*)&As[rA0][cA0] = *(const uint4*)&A[(size_t)(m0 + rA0) * K + kk + cA0];
        *(uint4*)&As[rA1][cA1] = *(const uint4*)&A[(size_t)(m0 + rA1) * K + kk + cA1];
        *(uint4*)&Bs[rB][cB]   = *(const uint4*)&B[(size_t)(n0 + rB) * K + kk + cB];
        if (kk + 32 < K) {  // hint the next tiles into cache
            __builtin_prefetch(&A[(size_t)(m0 + rA0) * K + kk + 32 + cA0], 0, 3);
            __builtin_prefetch(&B[(size_t)(n0 + rB) * K + kk + 32 + cB], 0, 3);
        }
        __syncthreads();

        Frag a0 = load_frag2(&As[wm + x][16 * g],      &As[wm + x][16 * g + 8]);
        Frag a1 = load_frag2(&As[wm + 16 + x][16 * g], &As[wm + 16 + x][16 * g + 8]);
        Frag b0 = load_frag2(&Bs[wn + x][16 * g],      &Bs[wn + x][16 * g + 8]);
        Frag b1 = load_frag2(&Bs[wn + 16 + x][16 * g], &Bs[wn + 16 + x][16 * g + 8]);

        c[0][0] = WMMA_BF16(a0.v, b0.v, c[0][0]);
        c[0][1] = WMMA_BF16(a0.v, b1.v, c[0][1]);
        c[1][0] = WMMA_BF16(a1.v, b0.v, c[1][0]);
        c[1][1] = WMMA_BF16(a1.v, b1.v, c[1][1]);
        __syncthreads();
    }

#pragma unroll
    for (int mt = 0; mt < 2; ++mt)
#pragma unroll
        for (int nt = 0; nt < 2; ++nt)
#pragma unroll
            for (int j = 0; j < 8; ++j) {
                int row = m0 + wm + mt * 16 + 8 * g + j;
                int col = n0 + wn + nt * 16 + x;
                if (OUT_BF16)
                    ((bf16_t*)Cout)[(size_t)row * N + col] = f2bf(c[mt][nt][j]);
                else
                    ((float*)Cout)[(size_t)row * N + col] = c[mt][nt][j];
            }
}

// ---------------------------------------------------------------------------
// 5) attention: one wave (32 thr) per (b, chunk, head).  No guarded loads:
//    K/V loaded at clamped positions; zero-padding re-applied later.
// ---------------------------------------------------------------------------
__global__ __launch_bounds__(32)
void attn_kernel(const bf16_t* __restrict__ qkv, const bf16_t* __restrict__ relb,
                 bf16_t* __restrict__ aout) {
    __shared__ float  AC[32][80];
    __shared__ float  BD[32][80];
    __shared__ bf16_t AT[32][96];
    __shared__ bf16_t Vs[96][64];

    const int lane = threadIdx.x;
    const int h    = blockIdx.x & 15;
    const int blk  = (blockIdx.x >> 4) & 255;
    const int b    = blockIdx.x >> 12;
    const int g = lane >> 4, x = lane & 15;
    const int base = blk * 32 - 31;                 // ctx j -> seq pos base
    const long long rowbase = (long long)b * 8192 + (long long)blk * 32;
    const long long bseq = (long long)b * 8192;

    // ---- stage V[96][64] with clamped rows (24 x b128 per lane) ----
#pragma unroll
    for (int t = 0; t < 24; ++t) {
        int ch = lane + 32 * t;
        int row = ch >> 3, col = (ch & 7) * 8;
        int pos = clampi(base + row, 0, 8191);
        *(uint4*)&Vs[row][col] =
            *(const uint4*)&qkv[(bseq + pos) * 3072 + 2048 + h * 64 + col];
    }

    // ---- A fragments for q (kept across all score tiles) ----
    Frag aq[2][2];
#pragma unroll
    for (int mt = 0; mt < 2; ++mt) {
        const bf16_t* qrow = qkv + (rowbase + mt * 16 + x) * 3072 + h * 64;
#pragma unroll
        for (int kk = 0; kk < 2; ++kk)
            aq[mt][kk] = load_frag2(qrow + 32 * kk + 16 * g, qrow + 32 * kk + 16 * g + 8);
    }

    // ---- scores: ac = q@k^T, bd = q@rel^T  -> LDS ----
    for (int nt = 0; nt < 5; ++nt) {
        int l = nt * 16 + x;
        int pos = clampi(base + l, 0, 8191);
        int lc = l > 70 ? 70 : l;
        const bf16_t* krow = qkv + (bseq + pos) * 3072 + 1024 + h * 64;
        const bf16_t* rrow = relb + (long long)lc * 1024 + h * 64;
        Frag bk[2], br[2];
#pragma unroll
        for (int kk = 0; kk < 2; ++kk) {
            bk[kk] = load_frag2(krow + 32 * kk + 16 * g, krow + 32 * kk + 16 * g + 8);
            br[kk] = load_frag2(rrow + 32 * kk + 16 * g, rrow + 32 * kk + 16 * g + 8);
        }
        v8f cac[2], cbd[2];
        cac[0] = zero8(); cac[1] = zero8();
        cbd[0] = zero8(); cbd[1] = zero8();
#pragma unroll
        for (int mt = 0; mt < 2; ++mt)
#pragma unroll
            for (int kk = 0; kk < 2; ++kk) {
                cac[mt] = WMMA_BF16(aq[mt][kk].v, bk[kk].v, cac[mt]);
                cbd[mt] = WMMA_BF16(aq[mt][kk].v, br[kk].v, cbd[mt]);
            }
#pragma unroll
        for (int mt = 0; mt < 2; ++mt)
#pragma unroll
            for (int j = 0; j < 8; ++j) {
                AC[mt * 16 + 8 * g + j][nt * 16 + x] = cac[mt][j];
                BD[mt * 16 + 8 * g + j][nt * 16 + x] = cbd[mt][j];
            }
    }
    __syncthreads();

    // ---- rel-shift + softcap + softmax: one query row per lane ----
    {
        const int i = lane;
        float mx = -3.0e38f;
        for (int jj = 0; jj < 71; ++jj) {
            int pos = base + jj;
            // zero-padded K => ac contribution 0 outside the sequence
            float acv = (pos >= 0 && pos < 8192) ? AC[i][jj] : 0.f;
            // exact reference _rel_shift (pad-1, flatten, crop) semantics:
            float bds;
            if (jj >= i)          bds = BD[i][jj - i];
            else if (jj == i - 1) bds = 0.f;
            else                  bds = BD[i - 1][72 - i + jj];
            float lg = tanhf((acv + bds) * 0.02f) * 50.f;
            AC[i][jj] = lg;
            mx = fmaxf(mx, lg);
        }
        float s = 0.f;
        for (int jj = 0; jj < 71; ++jj) {
            float e = __expf(AC[i][jj] - mx);
            AC[i][jj] = e;
            s += e;
        }
        float rs = 1.f / s;
        for (int jj = 0; jj < 71; ++jj) {
            int pos = base + jj;
            // zero-padded V => null the weight so clamped V rows contribute 0
            bool valid = (pos >= 0 && pos < 8192);
            AT[i][jj] = valid ? f2bf(AC[i][jj] * rs) : f2bf(0.f);
        }
        for (int jj = 71; jj < 96; ++jj) AT[i][jj] = f2bf(0.f);
    }
    __syncthreads();

    // ---- out[32x64] = attn[32x96] @ V[96x64] ----
    Frag aa[2][3];
#pragma unroll
    for (int mt = 0; mt < 2; ++mt)
#pragma unroll
        for (int kk = 0; kk < 3; ++kk)
            aa[mt][kk] = load_frag2(&AT[mt * 16 + x][32 * kk + 16 * g],
                                    &AT[mt * 16 + x][32 * kk + 16 * g + 8]);
#pragma unroll
    for (int nt = 0; nt < 4; ++nt) {
        int d = nt * 16 + x;
        Frag bv[3];
#pragma unroll
        for (int kk = 0; kk < 3; ++kk)
#pragma unroll
            for (int j = 0; j < 8; ++j) {
                int cix = 32 * kk + 16 * g + 2 * j;
                BfPair p;
                p.h[0] = Vs[cix][d];
                p.h[1] = Vs[cix + 1][d];
                bv[kk].u[j] = p.u;
            }
        v8f co[2];
        co[0] = zero8(); co[1] = zero8();
#pragma unroll
        for (int mt = 0; mt < 2; ++mt)
#pragma unroll
            for (int kk = 0; kk < 3; ++kk)
                co[mt] = WMMA_BF16(aa[mt][kk].v, bv[kk].v, co[mt]);
#pragma unroll
        for (int mt = 0; mt < 2; ++mt)
#pragma unroll
            for (int j = 0; j < 8; ++j) {
                long long gr = rowbase + mt * 16 + 8 * g + j;
                aout[gr * 1024 + h * 64 + d] = f2bf(co[mt][j]);
            }
    }
}

// ---------------------------------------------------------------------------
// Launcher. Workspace layout (bytes):
//   [0,          67108864)  hidden bf16   [32768][1024]
//   [67108864,   75497472)  weights bf16  [4096][1024]  (wq|wk|wv|wpost)
//   [75497472,   75642880)  rel_k bf16    [71][1024]
//   [75642880,  276969472)  qkv bf16      [32768][3072]
//   [276969472, 344078336)  attn out bf16 [32768][1024]
// ---------------------------------------------------------------------------
extern "C" void kernel_launch(void* const* d_in, const int* in_sizes, int n_in,
                              void* d_out, int out_size, void* d_ws, size_t ws_size,
                              hipStream_t stream) {
    const float* hid = (const float*)d_in[0];
    const float* pe  = (const float*)d_in[1];
    const float* wq  = (const float*)d_in[2];
    const float* wk  = (const float*)d_in[3];
    const float* wv  = (const float*)d_in[4];
    const float* wr  = (const float*)d_in[5];
    const float* wp  = (const float*)d_in[6];
    const float* pds = (const float*)d_in[7];
    float* out = (float*)d_out;

    char* ws = (char*)d_ws;
    bf16_t* hidb  = (bf16_t*)(ws);
    bf16_t* wb    = (bf16_t*)(ws + (size_t)67108864);
    bf16_t* relb  = (bf16_t*)(ws + (size_t)75497472);
    bf16_t* qkvb  = (bf16_t*)(ws + (size_t)75642880);
    bf16_t* aoutb = (bf16_t*)(ws + (size_t)276969472);

    hipLaunchKernelGGL(cvt_hidden, dim3(32768), dim3(256), 0, stream, hid, hidb);
    hipLaunchKernelGGL(cvt_weights, dim3(4096), dim3(256), 0, stream, wq, wk, wv, wp, pds, wb);
    hipLaunchKernelGGL(relk_kernel, dim3(284), dim3(256), 0, stream, pe, wr, relb);

    // QKV = hidden @ [wq|wk|wv]^T -> bf16
    hipLaunchKernelGGL((gemm_bf16_kernel<true>), dim3(256, 48), dim3(256), 0, stream,
                       hidb, wb, (void*)qkvb, 32768, 3072, 1024);

    // windowed attention: 1 wave per (b, chunk, head)
    hipLaunchKernelGGL(attn_kernel, dim3(16384), dim3(32), 0, stream, qkvb, relb, aoutb);

    // final projection -> f32 output
    hipLaunchKernelGGL((gemm_bf16_kernel<false>), dim3(256, 16), dim3(256), 0, stream,
                       aoutb, wb + (size_t)3072 * 1024, (void*)out, 32768, 1024, 1024);
}
